// SpatialAttention_89163521065224
// MI455X (gfx1250) — compile-verified, matrix operand order checked by reading, not run on hardware
//
#include <hip/hip_runtime.h>
#include <hip/hip_bf16.h>

// ---------------------------------------------------------------------------
// SpatialAttention for MI455X (gfx1250): bf16 WMMA pipeline + TDM staging
//   conv x -> bf16 ; transpose-conv weights -> bf16
//   GEMM(x,WqT)+RMSNorm -> Qb ; GEMM(x,WkT)+RMSNorm -> Kb ; GEMM(x,WvT) -> Vb
//   flash attention (softcap+softmax) -> Ab ; GEMM(Ab,WoT) -> out (fp32)
// GEMM/attention A/B/K tiles staged into LDS with tensor_load_to_lds (TDM).
// ---------------------------------------------------------------------------

typedef __attribute__((ext_vector_type(16))) __bf16 v16bf;
typedef __attribute__((ext_vector_type(8)))  float  f32x8;
typedef __attribute__((ext_vector_type(4)))  unsigned int u32x4;
typedef __attribute__((ext_vector_type(8)))  int i32x8;
typedef __attribute__((ext_vector_type(4)))  int i32x4;

constexpr int R   = 8 * 16 * 256;   // B*T*S = 32768 rows
constexpr int D   = 1024;
constexpr int H   = 16;
constexpr int HD  = 64;
constexpr int HN  = H * HD;         // 1024
constexpr int KN  = 4 * HD;         // KVH*HD = 256
constexpr int S   = 256;
constexpr float QK_SCALE = 0.125f;  // HD^-0.5
constexpr float CAP      = 50.0f;
constexpr float RMS_EPS  = 1e-6f;

#define DEVFN __device__ __forceinline__

DEVFN f32x8 wmma_bf16(v16bf a, v16bf b, f32x8 c) {
  return __builtin_amdgcn_wmma_f32_16x16x32_bf16(
      /*neg_a=*/false, a, /*neg_b=*/false, b,
      /*c_mod=*/(short)0, c, /*reuse_a=*/false, /*reuse_b=*/false);
}

union FragU { v16bf v; u32x4 q[2]; };
union VecU  { u32x4 q; __bf16 e[8]; };

// Load one 16x32 bf16 WMMA fragment (A layout; B identical when the source is
// stored n-major). Per-lane chunks: k = half*8..+7 and k = 16+half*8..+7.
DEVFN v16bf ldfrag(const __bf16* base, int stride, int row, int k0) {
  const int lane = threadIdx.x & 31;
  const int m = lane & 15, hh = lane >> 4;
  FragU u;
  const __bf16* p = base + (size_t)(row + m) * stride + (k0 + hh * 8);
  u.q[0] = *(const u32x4*)(p);
  u.q[1] = *(const u32x4*)(p + 16);
  return u.v;
}

DEVFN f32x8 zero8() { f32x8 z = {0.f,0.f,0.f,0.f,0.f,0.f,0.f,0.f}; return z; }

// ---------------------------------------------------------------------------
// Tensor Data Mover: 2-D bf16 tile (tile0 elems per row, tile1 rows) from
// global (row stride stride0 elems) into LDS, with LDS row padding.
// pad_icode: pad after 2^(pad_icode+1) DWORDs. pad_acode: pad (pad_acode+1) DWORDs.
// Issue from ONE full wave; follow with s_wait_tensorcnt + barrier.
DEVFN void tdm_load_2d(unsigned lds_off, const void* gptr,
                       unsigned tile0, unsigned tile1, unsigned stride0,
                       unsigned pad_icode, unsigned pad_acode) {
  unsigned long long ga = (unsigned long long)(uintptr_t)gptr;
  u32x4 g0;
  g0[0] = 1u;                                   // count=1, user descriptor
  g0[1] = lds_off;                              // D#.lds_addr (bytes)
  g0[2] = (unsigned)ga;                         // global_addr[31:0]
  g0[3] = (unsigned)((ga >> 32) & 0x1FFFFFFu) | (2u << 30);  // [56:32] | type=2
  i32x8 g1;
  g1[0] = (int)((1u << 16)                      // data_size = 1 -> 2 bytes
              | (1u << 20)                      // pad_enable
              | (pad_icode << 22)               // pad_interval
              | (pad_acode << 25));             // pad_amount
  g1[1] = (int)((tile0 & 0xFFFFu) << 16);       // abar=0 | tensor_dim0[15:0]
  g1[2] = (int)((tile0 >> 16) | ((tile1 & 0xFFFFu) << 16)); // dim0 hi | dim1 lo
  g1[3] = (int)((tile1 >> 16) | (tile0 << 16)); // dim1 hi | tile_dim0
  g1[4] = (int)(tile1 & 0xFFFFu);               // tile_dim1 | tile_dim2=0
  g1[5] = (int)stride0;                         // tensor_dim0_stride[31:0]
  g1[6] = 0;                                    // stride0 hi | dim1_stride lo
  g1[7] = 0;
  i32x4 z4 = {0, 0, 0, 0};                      // groups 2/3 unused (2-D)
  i32x8 z8 = {0, 0, 0, 0, 0, 0, 0, 0};
  __builtin_amdgcn_tensor_load_to_lds(g0, g1, z4, z4, z8, 0);
}

// ---------------------------------------------------------------------------
// elementwise fp32 -> bf16
__global__ void conv_bf16_k(const float* __restrict__ in, __bf16* __restrict__ out, int n) {
  for (int i = blockIdx.x * blockDim.x + threadIdx.x; i < n; i += gridDim.x * blockDim.x)
    out[i] = (__bf16)in[i];
}

// W (K,N) row-major fp32  ->  Wt (N,K) row-major bf16
__global__ void transpose_conv_k(const float* __restrict__ in, __bf16* __restrict__ out,
                                 int K, int N) {
  int total = K * N;
  for (int i = blockIdx.x * blockDim.x + threadIdx.x; i < total; i += gridDim.x * blockDim.x) {
    int k = i / N, n = i - k * N;
    out[(size_t)n * K + k] = (__bf16)in[i];
  }
}

// ---------------------------------------------------------------------------
// bf16 GEMM: C(M,N) = A(M,K) * Bt(N,K)^T, f32 accumulate.
// Block tile 128x128, 8 waves (4x2), wave tile 32x64, K step 64, TDM-staged LDS.
// MODE 0: fp32 out. MODE 1: bf16 out with fused per-64-col RMSNorm (weights nw).
// MODE 2: bf16 out plain.
template <int MODE>
__global__ __launch_bounds__(256)
void gemm_bf16_k(const __bf16* __restrict__ A, const __bf16* __restrict__ Bt,
                 void* __restrict__ Cout, int K, int N, const float* __restrict__ nw) {
  constexpr int LDS_STRIDE = 72;                 // 64 + 8 pad (144B rows, 16B aligned)
  __shared__ __bf16 sA[128 * LDS_STRIDE];
  __shared__ __bf16 sB[128 * LDS_STRIDE];

  const int tid  = threadIdx.x;
  const int w    = tid >> 5;
  const int wm   = w >> 1;                       // 0..3 (M)
  const int wn   = w & 1;                        // 0..1 (N)
  const int lane = tid & 31;
  const int m16  = lane & 15, hh = lane >> 4;
  const size_t rowBase = (size_t)blockIdx.y * 128;
  const int    colBase = blockIdx.x * 128;

  f32x8 acc[2][4];
#pragma unroll
  for (int i = 0; i < 2; i++)
#pragma unroll
    for (int j = 0; j < 4; j++) acc[i][j] = zero8();

  const unsigned sA_off = (unsigned)(uintptr_t)&sA[0];
  const unsigned sB_off = (unsigned)(uintptr_t)&sB[0];

  for (int kt = 0; kt < K; kt += 64) {
    __syncthreads();                             // previous round's reads done
    if (w == 0) {
      // 128x64 bf16 tiles; LDS rows padded 128B+16B (interval 32 DW, pad 4 DW)
      tdm_load_2d(sA_off, A  + rowBase * (size_t)K + kt, 64, 128, (unsigned)K, 4, 3);
      tdm_load_2d(sB_off, Bt + (size_t)colBase * (size_t)K + kt, 64, 128, (unsigned)K, 4, 3);
      __builtin_amdgcn_s_wait_tensorcnt(0);
    }
    __syncthreads();

#pragma unroll
    for (int kc = 0; kc < 2; kc++) {
      v16bf af[2], bfv[4];
#pragma unroll
      for (int i = 0; i < 2; i++) af[i] = ldfrag(sA, LDS_STRIDE, wm * 32 + i * 16, kc * 32);
#pragma unroll
      for (int j = 0; j < 4; j++) bfv[j] = ldfrag(sB, LDS_STRIDE, wn * 64 + j * 16, kc * 32);
#pragma unroll
      for (int i = 0; i < 2; i++)
#pragma unroll
        for (int j = 0; j < 4; j++) acc[i][j] = wmma_bf16(af[i], bfv[j], acc[i][j]);
    }
  }

  // Epilogue. C layout: elem e -> row = e + 8*hh (rel. to 16-row tile),
  // col = m16 within each 16-col tile. A wave's 64 cols == one head (RMSNorm).
#pragma unroll
  for (int i = 0; i < 2; i++) {
    if (MODE == 1) {
#pragma unroll
      for (int e = 0; e < 8; e++) {
        float ss = 0.f;
#pragma unroll
        for (int j = 0; j < 4; j++) { float v = acc[i][j][e]; ss += v * v; }
#pragma unroll
        for (int msk = 1; msk < 16; msk <<= 1) ss += __shfl_xor(ss, msk, 32);
        float sc = rsqrtf(ss * (1.0f / 64.0f) + RMS_EPS);
#pragma unroll
        for (int j = 0; j < 4; j++) acc[i][j][e] *= sc;
      }
    }
#pragma unroll
    for (int j = 0; j < 4; j++) {
      const int col = colBase + wn * 64 + j * 16 + m16;
      const float wgt = (MODE == 1) ? nw[j * 16 + m16] : 1.0f;  // col & 63
#pragma unroll
      for (int e = 0; e < 8; e++) {
        const size_t row = rowBase + wm * 32 + i * 16 + e + 8 * hh;
        float v = acc[i][j][e];
        if (MODE == 1) v *= wgt;
        if (MODE == 0) ((float*)Cout)[row * N + col] = v;
        else           ((__bf16*)Cout)[row * N + col] = (__bf16)v;
      }
    }
  }
}

// ---------------------------------------------------------------------------
// Flash attention with soft cap. One workgroup per (q-half, head, bt).
// 8 waves x 16 q-rows = 128 rows per block; S=256 keys in 4 blocks of 64.
// K tile staged by TDM; V transposed manually (TDM cannot transpose).
__global__ __launch_bounds__(256)
void attn_k(const __bf16* __restrict__ Qb, const __bf16* __restrict__ Kb,
            const __bf16* __restrict__ Vb, __bf16* __restrict__ Ab) {
  constexpr int LK = 72;                          // 64 + 8 pad
  __shared__ __bf16 sK [64 * LK];                 // [key][hd]
  __shared__ __bf16 sVt[64 * LK];                 // [hd][key]
  __shared__ __bf16 sP [8][16 * LK];              // per-wave P staging [row][key]

  const int tid  = threadIdx.x;
  const int w    = tid >> 5;
  const int lane = tid & 31;
  const int m16  = lane & 15, hh = lane >> 4;
  const int bt   = blockIdx.z;
  const int h    = blockIdx.y;
  const int kvh  = h >> 2;                        // GROUPS = 4
  const int qbase = blockIdx.x * 128 + w * 16;
  const size_t qrow0 = (size_t)bt * S + qbase;
  const unsigned sK_off = (unsigned)(uintptr_t)&sK[0];

  // Q fragments straight from global (A layout), HD=64 -> 2 K-chunks
  v16bf aq[2];
#pragma unroll
  for (int c = 0; c < 2; c++) {
    FragU u;
    const __bf16* p = Qb + (qrow0 + m16) * (size_t)HN + h * HD + c * 32 + hh * 8;
    u.q[0] = *(const u32x4*)p;
    u.q[1] = *(const u32x4*)(p + 16);
    aq[c] = u.v;
  }

  f32x8 o[4];
#pragma unroll
  for (int j = 0; j < 4; j++) o[j] = zero8();
  float mrun[8], lrun[8];
#pragma unroll
  for (int e = 0; e < 8; e++) { mrun[e] = -1e30f; lrun[e] = 0.f; }

  for (int kb = 0; kb < 4; kb++) {
    __syncthreads();
    const size_t blk0 = ((size_t)bt * S + kb * 64) * (size_t)KN + kvh * HD;
    if (w == 0) {                                 // TDM: 64x64 bf16 K tile
      tdm_load_2d(sK_off, Kb + blk0, 64, 64, (unsigned)KN, 4, 3);
    }
    // stage V transposed: [hd][key]
#pragma unroll
    for (int i = 0; i < 2; i++) {
      int idx = tid + i * 256;                    // 0..511 = 64 rows x 8 chunks
      int r = idx >> 3, ch = idx & 7;
      VecU uv; uv.q = *(const u32x4*)(Vb + blk0 + (size_t)r * KN + ch * 8);
#pragma unroll
      for (int jj = 0; jj < 8; jj++) sVt[(ch * 8 + jj) * LK + r] = uv.e[jj];
    }
    if (w == 0) __builtin_amdgcn_s_wait_tensorcnt(0);
    __syncthreads();

    // scores: S(16x64) = Q(16x64) * K^T ; 4 key tiles x 2 hd chunks
    f32x8 s[4];
#pragma unroll
    for (int j = 0; j < 4; j++) s[j] = zero8();
#pragma unroll
    for (int j = 0; j < 4; j++)
#pragma unroll
      for (int c = 0; c < 2; c++) {
        v16bf bk = ldfrag(sK, LK, j * 16, c * 32);
        s[j] = wmma_bf16(aq[c], bk, s[j]);
      }

    // scale + soft cap
#pragma unroll
    for (int j = 0; j < 4; j++)
#pragma unroll
      for (int e = 0; e < 8; e++) {
        float v = s[j][e] * QK_SCALE;
        s[j][e] = CAP * tanhf(v * (1.0f / CAP));
      }

    // online softmax; row r = e + 8*hh lives in one 16-lane half
#pragma unroll
    for (int e = 0; e < 8; e++) {
      float mb = -1e30f;
#pragma unroll
      for (int j = 0; j < 4; j++) mb = fmaxf(mb, s[j][e]);
#pragma unroll
      for (int msk = 1; msk < 16; msk <<= 1) mb = fmaxf(mb, __shfl_xor(mb, msk, 32));
      float mnew = fmaxf(mrun[e], mb);
      float f = __expf(mrun[e] - mnew);
      float ps = 0.f;
#pragma unroll
      for (int j = 0; j < 4; j++) {
        float p = __expf(s[j][e] - mnew);
        s[j][e] = p;
        ps += p;
      }
#pragma unroll
      for (int msk = 1; msk < 16; msk <<= 1) ps += __shfl_xor(ps, msk, 32);
      lrun[e] = lrun[e] * f + ps;
      mrun[e] = mnew;
#pragma unroll
      for (int j = 0; j < 4; j++) o[j][e] *= f;
    }

    // C-layout -> A-layout via wave-private LDS staging (bf16)
#pragma unroll
    for (int j = 0; j < 4; j++)
#pragma unroll
      for (int e = 0; e < 8; e++)
        sP[w][(e + 8 * hh) * LK + j * 16 + m16] = (__bf16)s[j][e];

    // O(16x64) += P(16x64) * V(64x64)
    v16bf ap[2];
#pragma unroll
    for (int c = 0; c < 2; c++) ap[c] = ldfrag(sP[w], LK, 0, c * 32);
#pragma unroll
    for (int j = 0; j < 4; j++)
#pragma unroll
      for (int c = 0; c < 2; c++) {
        v16bf bv = ldfrag(sVt, LK, j * 16, c * 32);
        o[j] = wmma_bf16(ap[c], bv, o[j]);
      }
  }

  // normalize and store bf16 into (bt*S+q, H*HD) layout for the O-proj GEMM
#pragma unroll
  for (int e = 0; e < 8; e++) {
    float inv = 1.0f / lrun[e];
    size_t row = qrow0 + e + 8 * hh;
#pragma unroll
    for (int j = 0; j < 4; j++)
      Ab[row * (size_t)HN + h * HD + j * 16 + m16] = (__bf16)(o[j][e] * inv);
  }
}

// ---------------------------------------------------------------------------
extern "C" void kernel_launch(void* const* d_in, const int* in_sizes, int n_in,
                              void* d_out, int out_size, void* d_ws, size_t ws_size,
                              hipStream_t stream) {
  (void)in_sizes; (void)n_in; (void)out_size; (void)ws_size;
  const float* x   = (const float*)d_in[0];
  const float* Wq  = (const float*)d_in[1];
  const float* Wk  = (const float*)d_in[2];
  const float* Wv  = (const float*)d_in[3];
  const float* Wo  = (const float*)d_in[4];
  const float* qnw = (const float*)d_in[5];
  const float* knw = (const float*)d_in[6];

  char* ws = (char*)d_ws;
  size_t off = 0;
  auto alloc = [&](size_t bytes) -> void* {
    void* p = (void*)(ws + off);
    off += (bytes + 255) & ~(size_t)255;
    return p;
  };

  __bf16* xb  = (__bf16*)alloc((size_t)R * D * 2);    // 67 MB (fits in 192MB L2)
  __bf16* WqT = (__bf16*)alloc((size_t)D * HN * 2);
  __bf16* WkT = (__bf16*)alloc((size_t)D * KN * 2);
  __bf16* WvT = (__bf16*)alloc((size_t)D * KN * 2);
  __bf16* WoT = (__bf16*)alloc((size_t)HN * D * 2);
  __bf16* Qb  = (__bf16*)alloc((size_t)R * HN * 2);   // RMS-normed q, bf16
  __bf16* Kb  = (__bf16*)alloc((size_t)R * KN * 2);   // RMS-normed k, bf16
  __bf16* Vb  = (__bf16*)alloc((size_t)R * KN * 2);
  __bf16* Ab  = (__bf16*)alloc((size_t)R * HN * 2);   // attention output, bf16

  conv_bf16_k<<<2048, 256, 0, stream>>>(x, xb, R * D);
  transpose_conv_k<<<1024, 256, 0, stream>>>(Wq, WqT, D, HN);
  transpose_conv_k<<<512,  256, 0, stream>>>(Wk, WkT, D, KN);
  transpose_conv_k<<<512,  256, 0, stream>>>(Wv, WvT, D, KN);
  transpose_conv_k<<<1024, 256, 0, stream>>>(Wo, WoT, HN, D);

  gemm_bf16_k<1><<<dim3(HN / 128, R / 128), 256, 0, stream>>>(xb, WqT, Qb, D, HN, qnw);
  gemm_bf16_k<1><<<dim3(KN / 128, R / 128), 256, 0, stream>>>(xb, WkT, Kb, D, KN, knw);
  gemm_bf16_k<2><<<dim3(KN / 128, R / 128), 256, 0, stream>>>(xb, WvT, Vb, D, KN, nullptr);

  attn_k<<<dim3(2, H, R / S), 256, 0, stream>>>(Qb, Kb, Vb, Ab);

  gemm_bf16_k<0><<<dim3(D / 128, R / 128), 256, 0, stream>>>(Ab, WoT, d_out, HN, D, nullptr);
}
